// HelixNeuralNetwork_80298708566059
// MI455X (gfx1250) — matrix operand order repaired and checked by reading, not compiled, not run on hardware
//
#include <hip/hip_runtime.h>
#include <math.h>

typedef __bf16 bf16;
typedef __attribute__((ext_vector_type(16))) __bf16 v16bf;
typedef __attribute__((ext_vector_type(8)))  __bf16 v8bf;
typedef __attribute__((ext_vector_type(8)))  float  v8f;

#define DT 0.1f

__device__ __forceinline__ v8f wmma_bf16(v16bf a, v16bf b, v8f c) {
    // D = A(16x32 bf16) x B(32x16 bf16) + C(f32)
    return __builtin_amdgcn_wmma_f32_16x16x32_bf16(false, a, false, b, (short)0, c, false, false);
}

__device__ __forceinline__ v16bf cat8(v8bf lo, v8bf hi) {
    return __builtin_shufflevector(lo, hi, 0,1,2,3,4,5,6,7,8,9,10,11,12,13,14,15);
}

// A-fragment (16x32 bf16): lane holds row m=lane%16; K halves {kb..kb+7, kb+16..kb+23}, kb=8*(lane/16)
__device__ __forceinline__ v16bf load_afrag(const bf16* base, int lane, int k, int ldk) {
    int m  = lane & 15;
    int kb = (lane >> 4) << 3;
    const v8bf* p0 = (const v8bf*)(base + m * ldk + k + kb);
    const v8bf* p1 = (const v8bf*)(base + m * ldk + k + kb + 16);
    return cat8(*p0, *p1);
}

// B-fragment (32x16 bf16): lane holds col n=lane%16; K halves {kb..kb+7, kb+16..kb+23}, kb=8*(lane/16).
// base points at element [n=0][k=0] of an n-major buffer with row stride ldk.
__device__ __forceinline__ v16bf load_bfrag(const bf16* base, int lane, int ldk) {
    const bf16* p = base + (lane & 15) * ldk + ((lane >> 4) << 3);
    const v8bf* p0 = (const v8bf*)p;
    const v8bf* p1 = (const v8bf*)(p + 16);
    return cat8(*p0, *p1);
}

__device__ __forceinline__ float gelu_f(float x) {
    return 0.5f * x * (1.0f + erff(x * 0.70710678118654752f));
}

__device__ __forceinline__ unsigned hashu(unsigned x) {
    x ^= x >> 16; x *= 0x7feb352dU; x ^= x >> 15; x *= 0x846ca68bU; x ^= x >> 16; return x;
}

// ---------------------------------------------------------------------------
// Fused Kuramoto layer: 10 steps, theta + its sin/cos register-resident per
// owning lane (WMMA D-tile ownership is step-invariant); sin/cos published to
// LDS in bf16 only for the A-fragments of the two GEMMs
//   (K sin th), (K cos th): coupling_i = cos_i*(K sin)_i - sin_i*(K cos)_i.
// K is symmetric -> B-fragments load contiguous rows of K straight from
// global (compiler keeps them VGPR-resident across all 10 steps).
// Block: 256 threads (8 waves). Wave w owns output columns [32w, 32w+32).
// ---------------------------------------------------------------------------
__global__ __launch_bounds__(256) void kuramoto_kernel(
    float* __restrict__ theta_g, const bf16* __restrict__ Kl,
    const float* __restrict__ omega, const float* __restrict__ om_add,
    const float* __restrict__ state, const float* __restrict__ Kg_ptr,
    float* __restrict__ r_out)
{
    __shared__ __align__(16) bf16  Sl[16 * 256];
    __shared__ __align__(16) bf16  Cl[16 * 256];
    __shared__ float red[2 * 256];

    const int tid  = threadIdx.x;
    const int lane = tid & 31;
    const int wv   = tid >> 5;
    const int row0 = blockIdx.x * 16;
    const int nb   = wv * 32;

    const float k_mult = state[1];
    const float coeff  = DT * Kg_ptr[0] * k_mult * (1.0f / 256.0f);

    // Per-lane owned elements: (m = e + 8*(lane/16), n = nb + 16*t + lane%16)
    const int mh = (lane >> 4) << 3;
    const int nl = lane & 15;

    float thr[2][8];   // theta, register resident
    float sReg[2][8];  // sin(theta)
    float cReg[2][8];  // cos(theta)
    float om[2];       // DT*(omega[n] + om_add[n]) per owned column

    #pragma unroll
    for (int t = 0; t < 2; ++t) {
        int n = nb + 16 * t + nl;
        om[t] = DT * (omega[n] + om_add[n]);
        #pragma unroll
        for (int e = 0; e < 8; ++e) {
            int m = e + mh;
            float tv = theta_g[(row0 + m) * 256 + n];
            thr[t][e] = tv;
            float s, c; __sincosf(tv, &s, &c);
            sReg[t][e] = s; cReg[t][e] = c;
            Sl[m * 256 + n] = (bf16)s;
            Cl[m * 256 + n] = (bf16)c;
        }
    }
    __syncthreads();

    for (int step = 0; step < 10; ++step) {
        v8f accS[2], accC[2];
        #pragma unroll
        for (int t = 0; t < 2; ++t) {
            accS[t] = (v8f){0.f,0.f,0.f,0.f,0.f,0.f,0.f,0.f};
            accC[t] = (v8f){0.f,0.f,0.f,0.f,0.f,0.f,0.f,0.f};
        }
        #pragma unroll
        for (int kc = 0; kc < 8; ++kc) {
            const int k = kc * 32;
            v16bf aS = load_afrag(Sl, lane, k, 256);
            v16bf aC = load_afrag(Cl, lane, k, 256);
            #pragma unroll
            for (int t = 0; t < 2; ++t) {
                // K symmetric: column n of K == row n of K -> contiguous loads,
                // loop-invariant across steps (compiler holds them in VGPRs)
                v16bf bF = load_bfrag(Kl + (nb + 16 * t) * 256 + k, lane, 256);
                accS[t] = wmma_bf16(aS, bF, accS[t]);
                accC[t] = wmma_bf16(aC, bF, accC[t]);
            }
        }
        __syncthreads(); // all A-fragment reads done before sin/cos rewrite

        #pragma unroll
        for (int t = 0; t < 2; ++t) {
            #pragma unroll
            for (int e = 0; e < 8; ++e) {
                float coup = cReg[t][e] * accS[t][e] - sReg[t][e] * accC[t][e];
                float tn = thr[t][e] + om[t] + coeff * coup;
                tn -= 6.28318530717958647f * rintf(tn * 0.15915494309189535f); // wrap (-pi,pi]
                thr[t][e] = tn;
                float s2, c2; __sincosf(tn, &s2, &c2);
                sReg[t][e] = s2; cReg[t][e] = c2;
                int m = e + mh, n = nb + 16 * t + nl;
                Sl[m * 256 + n] = (bf16)s2;
                Cl[m * 256 + n] = (bf16)c2;
            }
        }
        __syncthreads();
    }

    // order parameter r[m] = |mean(e^{i theta})| over 256 oscillators
    {
        int m = tid >> 4, cg = tid & 15;
        float sc = 0.f, ss = 0.f;
        #pragma unroll
        for (int j = 0; j < 16; ++j) {
            sc += (float)Cl[m * 256 + cg * 16 + j];
            ss += (float)Sl[m * 256 + cg * 16 + j];
        }
        red[tid] = sc; red[256 + tid] = ss;
    }
    __syncthreads();
    if (tid < 16) {
        float c = 0.f, s = 0.f;
        #pragma unroll
        for (int j = 0; j < 16; ++j) { c += red[tid * 16 + j]; s += red[256 + tid * 16 + j]; }
        c *= (1.0f / 256.0f); s *= (1.0f / 256.0f);
        r_out[row0 + tid] = sqrtf(c * c + s * s);
    }
    // write back owned theta
    #pragma unroll
    for (int t = 0; t < 2; ++t) {
        int n = nb + 16 * t + nl;
        #pragma unroll
        for (int e = 0; e < 8; ++e)
            theta_g[(row0 + e + mh) * 256 + n] = thr[t][e];
    }
}

// ---------------------------------------------------------------------------
// Generic WMMA GEMM: C[M,N] = act(A[M,K] @ B[K,N] + bias), f32 in/out, bf16 MAC.
// Block 256 threads = 8 waves; block tile 64x64; wave w -> tiles {2w, 2w+1}.
// act: 0=none, 1=gelu, 2=tanh*pi
// ---------------------------------------------------------------------------
__global__ __launch_bounds__(256) void gemm_bf16_kernel(
    const float* __restrict__ A, const float* __restrict__ B,
    const float* __restrict__ bias, float* __restrict__ C,
    int M, int N, int K, int act)
{
    __shared__ __align__(16) bf16 Ab[64 * 32];
    __shared__ __align__(16) bf16 Bt[64 * 32];

    const int tid  = threadIdx.x;
    const int lane = tid & 31;
    const int wv   = tid >> 5;
    const int mB = blockIdx.x * 64, nB = blockIdx.y * 64;
    const int rowg = wv >> 1;

    v8f acc[2];
    acc[0] = (v8f){0.f,0.f,0.f,0.f,0.f,0.f,0.f,0.f};
    acc[1] = (v8f){0.f,0.f,0.f,0.f,0.f,0.f,0.f,0.f};

    for (int k0 = 0; k0 < K; k0 += 32) {
        #pragma unroll
        for (int i = 0; i < 8; ++i) { // A chunk 64x32, row-major bf16
            int idx = tid + i * 256;
            int r = idx >> 5, c = idx & 31;
            Ab[r * 32 + c] = (bf16)A[(mB + r) * K + k0 + c];
        }
        #pragma unroll
        for (int i = 0; i < 8; ++i) { // B chunk 32x64 -> transposed (n-major) bf16
            int idx = tid + i * 256;
            int kl = idx >> 6, nl = idx & 63;
            Bt[nl * 32 + kl] = (bf16)B[(k0 + kl) * N + nB + nl];
        }
        __syncthreads();

        v16bf aF = load_afrag(Ab + rowg * 16 * 32, lane, 0, 32);
        #pragma unroll
        for (int j = 0; j < 2; ++j) {
            int colg = 2 * (wv & 1) + j;
            v16bf bF = load_bfrag(Bt + colg * 16 * 32, lane, 32);
            acc[j] = wmma_bf16(aF, bF, acc[j]);
        }
        __syncthreads();
    }

    #pragma unroll
    for (int j = 0; j < 2; ++j) {
        int colg = 2 * (wv & 1) + j;
        int n = nB + colg * 16 + (lane & 15);
        float bv = bias ? bias[n] : 0.f;
        #pragma unroll
        for (int e = 0; e < 8; ++e) {
            int m = mB + rowg * 16 + e + ((lane >> 4) << 3);
            float v = acc[j][e] + bv;
            if (act == 1) v = gelu_f(v);
            else if (act == 2) v = tanhf(v) * 3.14159265358979323f;
            C[m * N + n] = v;
        }
    }
}

// ---------------------------------------------------------------------------
// Sinkhorn normalization (log-space), 256x256
// ---------------------------------------------------------------------------
__global__ __launch_bounds__(256) void sinkhorn_row_kernel(
    const float* __restrict__ src, float* __restrict__ dst)
{
    __shared__ float red[256];
    int r = blockIdx.x, t = threadIdx.x;
    float v = src[r * 256 + t];
    red[t] = v; __syncthreads();
    for (int s = 128; s > 0; s >>= 1) { if (t < s) red[t] = fmaxf(red[t], red[t + s]); __syncthreads(); }
    float mx = red[0]; __syncthreads();
    red[t] = __expf(v - mx); __syncthreads();
    for (int s = 128; s > 0; s >>= 1) { if (t < s) red[t] += red[t + s]; __syncthreads(); }
    dst[r * 256 + t] = v - (mx + __logf(red[0]));
}

__global__ __launch_bounds__(256) void sinkhorn_col_kernel(
    const float* __restrict__ src, float* __restrict__ dst, int do_exp)
{
    __shared__ float red[256];
    int c = blockIdx.x, t = threadIdx.x;
    float v = src[t * 256 + c];
    red[t] = v; __syncthreads();
    for (int s = 128; s > 0; s >>= 1) { if (t < s) red[t] = fmaxf(red[t], red[t + s]); __syncthreads(); }
    float mx = red[0]; __syncthreads();
    red[t] = __expf(v - mx); __syncthreads();
    for (int s = 128; s > 0; s >>= 1) { if (t < s) red[t] += red[t + s]; __syncthreads(); }
    float o = v - (mx + __logf(red[0]));
    dst[t * 256 + c] = do_exp ? __expf(o) : o;
}

// ---------------------------------------------------------------------------
// Small state kernels
// ---------------------------------------------------------------------------
__global__ __launch_bounds__(256) void kconv_kernel(const float* __restrict__ K, bf16* __restrict__ K16)
{
    int i = blockIdx.x * 256 + threadIdx.x;
    K16[i] = (bf16)K[i];
}

// state: [0]=z [1]=k_mult [2]=sm [3]=op [4]=rmean
__global__ __launch_bounds__(256) void init_kernel(float* state, float* om_add, float* lsum)
{
    int t = threadIdx.x;
    om_add[t] = 0.f;
    if (t == 0) { state[0] = 0.1f; state[1] = 1.f; state[2] = 0.f; state[3] = 0.f; state[4] = 0.f; }
    if (t < 6) lsum[t] = 0.f;
}

__global__ __launch_bounds__(256) void zstep_kernel(
    const float* __restrict__ r, float* state, float* lsum,
    const float* __restrict__ zm_p, const float* __restrict__ zd_p)
{
    __shared__ float red[256];
    int t = threadIdx.x;
    float s = 0.f;
    for (int i = t; i < 1024; i += 256) s += r[i];
    red[t] = s; __syncthreads();
    for (int st = 128; st > 0; st >>= 1) { if (t < st) red[t] += red[t + st]; __syncthreads(); }
    if (t == 0) {
        float rmean = red[0] * (1.0f / 1024.0f);
        float z = state[0];
        float dz = zm_p[0] * (rmean - z) - zd_p[0] * (z - 0.5f);
        z = fminf(fmaxf(z + 0.01f * dz, 0.f), 1.f);
        state[0] = z; state[4] = rmean;
    }
    if (t < 6) lsum[t] = 0.f; // reset logit accumulator for this layer's selector
}

__global__ __launch_bounds__(64) void selector_kernel(
    const float* __restrict__ theta, const float* __restrict__ r,
    const float* __restrict__ state,
    const float* __restrict__ W1, const float* __restrict__ b1,
    const float* __restrict__ W2, const float* __restrict__ b2,
    float* __restrict__ lsum)
{
    __shared__ float h[64];
    int row = blockIdx.x, j = threadIdx.x;
    float z = state[0];
    float acc = b1[j];
    const float* tr = theta + row * 256;
    for (int k = 0; k < 256; ++k) acc += tr[k] * W1[k * 64 + j];
    acc += r[row] * W1[256 * 64 + j] + z * W1[257 * 64 + j];
    h[j] = gelu_f(acc);
    __syncthreads();
    if (j < 6) {
        float l = b2[j];
        #pragma unroll
        for (int i = 0; i < 64; ++i) l += h[i] * W2[i * 6 + j];
        atomicAdd(&lsum[j], l);
    }
}

__global__ void opsel_kernel(const float* __restrict__ lsum,
                             const float* __restrict__ op_strength, float* state)
{
    float best = lsum[0]; int op = 0;
    for (int i = 1; i < 6; ++i) { float v = lsum[i]; if (v > best) { best = v; op = i; } }
    float sm = op_strength[op] * state[4];
    state[1] = (op == 1) ? (1.f + sm * 0.5f) : ((op == 2) ? (1.f - sm * 0.3f) : 1.f);
    state[2] = sm;
    state[3] = (float)op;
}

__global__ __launch_bounds__(256) void omadd_kernel(
    const float* __restrict__ state, const float* __restrict__ catalyze, float* om_add)
{
    int t = threadIdx.x;
    int op = (int)state[3];
    om_add[t] = (op == 5) ? catalyze[t] * state[2] : 0.f;
}

__global__ __launch_bounds__(256) void apply_op_kernel(
    float* __restrict__ theta, const float* __restrict__ theta_ex,
    const float* __restrict__ r, const float* __restrict__ state,
    const float* __restrict__ op_strength, int li)
{
    int i = blockIdx.x * 256 + threadIdx.x;
    int b = i >> 8;
    int op = (int)state[3];
    float t = theta[i];
    if (op == 3) {
        t = theta_ex[i];                       // EXCHANGE: theta @ P
    } else if (op == 4) {                      // INHIBIT: gaussian noise (hash-based)
        unsigned h1 = hashu((unsigned)i * 2654435761u + (unsigned)li * 0x9E3779B9u + 0x42u);
        unsigned h2 = hashu(h1 + 0x632BE59Bu);
        float u1 = (float)(h1 >> 8) * (1.0f / 16777216.0f) + 1e-7f;
        float u2 = (float)(h2 >> 8) * (1.0f / 16777216.0f);
        float nz = sqrtf(-2.f * __logf(u1)) * __cosf(6.28318530718f * u2);
        t += nz * (op_strength[4] * r[b]) * 0.2f;
    }
    theta[i] = t;
}

__global__ __launch_bounds__(256) void emb_kernel(const float* __restrict__ theta, float* __restrict__ emb)
{
    int b = blockIdx.x, j = threadIdx.x;
    float t = theta[b * 256 + j];
    emb[b * 512 + j]       = __cosf(t);
    emb[b * 512 + 256 + j] = __sinf(t);
}

// ---------------------------------------------------------------------------
extern "C" void kernel_launch(void* const* d_in, const int* in_sizes, int n_in,
                              void* d_out, int out_size, void* d_ws, size_t ws_size,
                              hipStream_t stream)
{
    (void)in_sizes; (void)n_in; (void)out_size; (void)ws_size;
    const float* x     = (const float*)d_in[0];
    const float* encW1 = (const float*)d_in[1];
    const float* encb1 = (const float*)d_in[2];
    const float* encW2 = (const float*)d_in[3];
    const float* encb2 = (const float*)d_in[4];
    const float* Kmat  = (const float*)d_in[5];
    const float* omega = (const float*)d_in[6];
    const float* Kglob = (const float*)d_in[7];
    const float* opstr = (const float*)d_in[8];
    const float* exlog = (const float*)d_in[9];
    const float* cata  = (const float*)d_in[10];
    const float* selW1 = (const float*)d_in[11];
    const float* selb1 = (const float*)d_in[12];
    const float* selW2 = (const float*)d_in[13];
    const float* selb2 = (const float*)d_in[14];
    const float* decW1 = (const float*)d_in[15];
    const float* decb1 = (const float*)d_in[16];
    const float* decW2 = (const float*)d_in[17];
    const float* decb2 = (const float*)d_in[18];
    const float* zm    = (const float*)d_in[19];
    const float* zd    = (const float*)d_in[20];

    char* ws = (char*)d_ws;
    size_t off = 0;
    auto alloc = [&](size_t bytes) -> char* {
        char* p = ws + off; off = (off + bytes + 255) & ~(size_t)255; return p;
    };
    float* theta = (float*)alloc(1024 * 256 * 4);
    float* hbuf  = (float*)alloc(1024 * 512 * 4);
    float* embb  = (float*)alloc(1024 * 512 * 4);
    float* gbuf  = (float*)alloc(1024 * 256 * 4);
    float* thex  = (float*)alloc(1024 * 256 * 4);
    float* Pws   = (float*)alloc(256 * 256 * 4);
    float* rbuf  = (float*)alloc(1024 * 4);
    float* omadd = (float*)alloc(256 * 4);
    float* state = (float*)alloc(64);
    float* lsum  = (float*)alloc(64);
    bf16*  K16   = (bf16*)alloc(4 * 256 * 256 * 2);

    kconv_kernel<<<1024, 256, 0, stream>>>(Kmat, K16);
    init_kernel<<<1, 256, 0, stream>>>(state, omadd, lsum);

    // encoder: gelu(x@W1+b1) then tanh(.)@pi
    gemm_bf16_kernel<<<dim3(16, 8), 256, 0, stream>>>(x,    encW1, encb1, hbuf,  1024, 512, 512, 1);
    gemm_bf16_kernel<<<dim3(16, 4), 256, 0, stream>>>(hbuf, encW2, encb2, theta, 1024, 256, 512, 2);

    // Sinkhorn soft permutation (5 iters, exp fused into last col pass)
    sinkhorn_row_kernel<<<256, 256, 0, stream>>>(exlog, Pws);
    sinkhorn_col_kernel<<<256, 256, 0, stream>>>(Pws, Pws, 0);
    for (int it = 1; it < 5; ++it) {
        sinkhorn_row_kernel<<<256, 256, 0, stream>>>(Pws, Pws);
        sinkhorn_col_kernel<<<256, 256, 0, stream>>>(Pws, Pws, (it == 4) ? 1 : 0);
    }

    for (int li = 0; li < 4; ++li) {
        kuramoto_kernel<<<64, 256, 0, stream>>>(theta, K16 + li * 65536, omega + li * 256,
                                                omadd, state, Kglob + li, rbuf);
        zstep_kernel<<<1, 256, 0, stream>>>(rbuf, state, lsum, zm, zd);
        selector_kernel<<<1024, 64, 0, stream>>>(theta, rbuf, state, selW1, selb1, selW2, selb2, lsum);
        opsel_kernel<<<1, 1, 0, stream>>>(lsum, opstr, state);
        // EXCHANGE candidate always computed (deterministic work), then select per op
        gemm_bf16_kernel<<<dim3(16, 4), 256, 0, stream>>>(theta, Pws, nullptr, thex, 1024, 256, 256, 0);
        apply_op_kernel<<<1024, 256, 0, stream>>>(theta, thex, rbuf, state, opstr, li);
        omadd_kernel<<<1, 256, 0, stream>>>(state, cata, omadd);
    }

    // decoder
    emb_kernel<<<1024, 256, 0, stream>>>(theta, embb);
    gemm_bf16_kernel<<<dim3(16, 4), 256, 0, stream>>>(embb, decW1, decb1, gbuf, 1024, 256, 512, 1);
    gemm_bf16_kernel<<<dim3(16, 2), 256, 0, stream>>>(gbuf, decW2, decb2, (float*)d_out, 1024, 128, 256, 0);
}
